// SlotAttentionAutoEncoder_1202590843742
// MI455X (gfx1250) — matrix operand order repaired and checked by reading, not compiled, not run on hardware
//
#include <hip/hip_runtime.h>
#include <hip/hip_bf16.h>
#include <math.h>

// ---------------------------------------------------------------------------
// SlotAttention autoencoder on MI455X (gfx1250, wave32).
// All GEMM-shaped work (9x 5x5 convs/deconvs, 1x 3x3 conv, 4x 64x64 linears)
// runs through one implicit-GEMM kernel built on v_wmma_f32_16x16x32_f16:
//   wave tile = 16 output pixels (M) x 64 output channels (4 N-tiles),
//   K = Cin = 64 per tap split into two 32-wide f16 chunks, f32 accumulate.
// Weights are pre-packed into the WMMA B-fragment lane layout and staged
// per-tap into LDS with double-buffered GLOBAL_LOAD_ASYNC_TO_LDS_B128
// (ASYNCcnt-tracked DMA), so tile fetch latency hides behind the WMMAs.
// ---------------------------------------------------------------------------

typedef __attribute__((ext_vector_type(16))) _Float16 v16h;
typedef __attribute__((ext_vector_type(8)))  _Float16 v8h;
typedef __attribute__((ext_vector_type(8)))  float    v8f;

#define LN_EPSF 1e-5f

// ============================ weight packing ===============================
// dst half index = (((tap*2 + kc)*ntiles + nt)*32 + lane)*16 + h
// value = w[o = nt*16 + (lane&15)][i = kc*32 + ((lane&16)?16:0) + h][ky][kx]
__global__ void pack_w_kernel(const float* __restrict__ src, _Float16* __restrict__ dst,
                              int Osrc, int Isrc, int KK, int ntiles)
{
    const int total = KK * KK * 2 * ntiles * 512;
    for (int idx = blockIdx.x * blockDim.x + threadIdx.x; idx < total;
         idx += gridDim.x * blockDim.x) {
        int h  = idx & 15;
        int t2 = idx >> 4;
        int lane = t2 & 31; t2 >>= 5;
        int nt = t2 % ntiles; t2 /= ntiles;
        int kc = t2 & 1; t2 >>= 1;
        int tap = t2;
        int o = nt * 16 + (lane & 15);
        int i = kc * 32 + ((lane & 16) ? 16 : 0) + h;
        float val = 0.f;
        if (o < Osrc && i < Isrc) {
            int ky = tap / KK, kx = tap % KK;
            val = src[(((size_t)o * Isrc + i) * KK + ky) * KK + kx];
        }
        dst[idx] = (_Float16)val;
    }
}

// image NCHW f32 (4,3,112,112) -> NHWC f16 with C zero-padded to 64
__global__ void pack_image_kernel(const float* __restrict__ src, _Float16* __restrict__ dst)
{
    const int total = 4 * 112 * 112 * 64;
    int idx = blockIdx.x * blockDim.x + threadIdx.x;
    if (idx >= total) return;
    int c = idx & 63;
    int p = idx >> 6;              // (b*112 + y)*112 + x
    int b = p / 12544, r = p - b * 12544;
    float v = 0.f;
    if (c < 3) v = src[((size_t)(b * 3 + c)) * 12544 + r];
    dst[idx] = (_Float16)v;
}

// ====================== implicit-GEMM WMMA conv ============================
__global__ __launch_bounds__(128)
void conv_wmma_kernel(const _Float16* __restrict__ in,
                      const _Float16* __restrict__ wpack,
                      const float* __restrict__ bias,
                      _Float16* __restrict__ out,
                      float* __restrict__ fout,
                      int Nimg, int inH, int inW, int outH, int outW,
                      int KK, int S, int pad, int Cout, int ntiles, int relu)
{
    // two 8KB buffers for double-buffered async weight staging
    __shared__ _Float16 __attribute__((aligned(32))) lb[2][4096];
    const int tid  = threadIdx.x;
    const int lane = tid & 31;
    const int wave = tid >> 5;
    const int totalPix = Nimg * outH * outW;
    const int gpix = (blockIdx.x * 4 + wave) * 16;

    // A-row (pixel) for this lane, per WMMA 16-bit A layout
    const int m = lane & 15;
    const int pix = gpix + m;
    const bool mvalid = pix < totalPix;
    const int cpix = mvalid ? pix : (totalPix - 1);
    const int img = cpix / (outH * outW);
    const int rem = cpix - img * outH * outW;
    const int oy  = rem / outW;
    const int ox  = rem - oy * outW;
    const int kb  = (lane & 16) ? 16 : 0;  // K-half base within a 32-chunk

    v8f zero8 = {0.f, 0.f, 0.f, 0.f, 0.f, 0.f, 0.f, 0.f};
    v8f acc[4]; acc[0] = zero8; acc[1] = zero8; acc[2] = zero8; acc[3] = zero8;
    v8h zh; for (int i = 0; i < 8; ++i) zh[i] = (_Float16)0.f;

    const int tapHalves   = 2 * ntiles * 512;          // halves per tap tile
    const int sliceBytes  = 512 * ntiles;              // bytes staged per wave
    const int T           = KK * KK;
    // LDS aperture keeps the byte offset in addr[31:0] (ISA 10.2) — truncate.
    const unsigned ldsBase = (unsigned)(size_t)(void*)&lb[0][0];

    // issue this wave's share (ntiles x B128 DMA) of tap's weight tile
    auto stage = [&](int buf, int tap) {
        unsigned gbase = (unsigned)tap * (unsigned)tapHalves * 2u + (unsigned)wave * sliceBytes;
        unsigned lbase = ldsBase + (unsigned)buf * 8192u + (unsigned)wave * sliceBytes;
        for (int q = 0; q < ntiles; ++q) {
            unsigned go = gbase + q * 512 + lane * 16;
            unsigned lo = lbase + q * 512 + lane * 16;
            asm volatile("global_load_async_to_lds_b128 %0, %1, %2"
                         :: "v"(lo), "v"(go), "s"(wpack) : "memory");
        }
    };

    stage(0, 0);
    for (int tap = 0; tap < T; ++tap) {
        const int cur = tap & 1;
        const bool hasNext = (tap + 1) < T;
        if (hasNext) stage(cur ^ 1, tap + 1);
        // async loads complete in order: allowing only the just-issued loads to
        // remain outstanding guarantees the current tap's tile has landed.
        if (hasNext) {
            if (ntiles == 4) asm volatile("s_wait_asynccnt 4" ::: "memory");
            else             asm volatile("s_wait_asynccnt 1" ::: "memory");
        } else {
            asm volatile("s_wait_asynccnt 0" ::: "memory");
        }
        __syncthreads();                       // tile visible to all 4 waves

        const int ky = tap / KK, kx = tap % KK;
        const int ty = oy - pad + ky;
        const bool vy = (ty >= 0) && ((ty % S) == 0) && ((ty / S) < inH);
        const int iy = vy ? (ty / S) : 0;
        const int tx = ox - pad + kx;
        const bool vx = (tx >= 0) && ((tx % S) == 0) && ((tx / S) < inW);
        const int ix = vx ? (tx / S) : 0;
        const bool valid = mvalid && vy && vx;
        const _Float16* arow = in + (((size_t)img * inH + iy) * inW + ix) * 64;
        const _Float16* lbc  = &lb[cur][0];
        for (int kc = 0; kc < 2; ++kc) {
            v8h lo = *(const v8h*)(arow + kc * 32 + kb);
            v8h hi = *(const v8h*)(arow + kc * 32 + kb + 8);
            lo = valid ? lo : zh;
            hi = valid ? hi : zh;
            v16h a = __builtin_shufflevector(lo, hi, 0, 1, 2, 3, 4, 5, 6, 7,
                                             8, 9, 10, 11, 12, 13, 14, 15);
            for (int nt = 0; nt < ntiles; ++nt) {
                const v16h bfrag = *(const v16h*)(lbc + (((kc * ntiles) + nt) * 32 + lane) * 16);
                acc[nt] = __builtin_amdgcn_wmma_f32_16x16x32_f16(
                    false, a, false, bfrag, (short)0, acc[nt], false, false);
            }
        }
        __syncthreads();   // all waves done reading lb[cur] before it is re-staged
    }

    // epilogue: D layout -> lane holds column (lane&15), rows j + 8*(lane>=16)
    const int nrow0 = (lane & 16) ? 8 : 0;
    const int chlo  = lane & 15;
    for (int nt = 0; nt < ntiles; ++nt) {
        const int ch = nt * 16 + chlo;
        if (ch >= Cout) continue;
        const float bv = bias[ch];
        for (int j = 0; j < 8; ++j) {
            const int p = gpix + nrow0 + j;
            if (p >= totalPix) continue;
            float val = acc[nt][j] + bv;
            if (relu) val = fmaxf(val, 0.f);
            const int im = p / (outH * outW);
            const int r2 = p - im * outH * outW;
            const int y = r2 / outW;
            const int x = r2 - y * outW;
            if (out) {
                out[(((size_t)im * outH + y) * outW + x) * 64 + ch] = (_Float16)val;
            } else if (y < 112 && x < 112) {
                if (ch < 3)       // recons (b,ns,3,112,112)
                    fout[((size_t)im * 3 + ch) * 12544 + y * 112 + x] = val;
                else if (ch == 3) // masks (b,ns,1,112,112)
                    fout[1053696 + (size_t)im * 12544 + y * 112 + x] = val;
            }
        }
    }
}

// ======================= encoder epilogue pieces ===========================
__global__ void add_pos_kernel(_Float16* __restrict__ act,
                               const float* __restrict__ w, const float* __restrict__ b)
{
    const int total = 4 * 12544 * 64;
    int idx = blockIdx.x * blockDim.x + threadIdx.x;
    if (idx >= total) return;
    int c = idx & 63;
    int p = (idx >> 6) % 12544;
    int i = p / 112, j = p - i * 112;
    float gy = i * (1.f / 111.f), gx = j * (1.f / 111.f);
    float pos = w[c * 4] * gy + w[c * 4 + 1] * gx +
                w[c * 4 + 2] * (1.f - gy) + w[c * 4 + 3] * (1.f - gx) + b[c];
    act[idx] = (_Float16)((float)act[idx] + pos);
}

__global__ void gstats_kernel(const _Float16* __restrict__ act, float* __restrict__ stats)
{
    const int per = 12544 * 64;
    int b = blockIdx.x >> 6, chunk = blockIdx.x & 63;
    float s = 0.f, ss = 0.f;
    for (int i = chunk * 256 + threadIdx.x; i < per; i += 64 * 256) {
        float v = (float)act[(size_t)b * per + i];
        s += v; ss += v * v;
    }
    __shared__ float r1[256], r2[256];
    r1[threadIdx.x] = s; r2[threadIdx.x] = ss; __syncthreads();
    for (int o = 128; o > 0; o >>= 1) {
        if (threadIdx.x < o) { r1[threadIdx.x] += r1[threadIdx.x + o]; r2[threadIdx.x] += r2[threadIdx.x + o]; }
        __syncthreads();
    }
    if (threadIdx.x == 0) { atomicAdd(&stats[b * 2], r1[0]); atomicAdd(&stats[b * 2 + 1], r2[0]); }
}

__global__ void gnorm_kernel(_Float16* __restrict__ act, const float* __restrict__ stats)
{
    const int per = 12544 * 64, total = 4 * per;
    int idx = blockIdx.x * blockDim.x + threadIdx.x;
    if (idx >= total) return;
    int b = idx / per;
    float n = (float)per;
    float mmean = stats[2 * b] / n;
    float var = stats[2 * b + 1] / n - mmean * mmean;
    float r = rsqrtf(var + LN_EPSF);
    act[idx] = (_Float16)(((float)act[idx] - mmean) * r);
}

__global__ void ln_rows_kernel(const _Float16* __restrict__ in, _Float16* __restrict__ out,
                               const float* __restrict__ g, const float* __restrict__ bb)
{
    int row = blockIdx.x, t = threadIdx.x;   // 64 threads
    float x = (float)in[(size_t)row * 64 + t];
    __shared__ float r1[64], r2[64];
    r1[t] = x; r2[t] = x * x; __syncthreads();
    for (int o = 32; o > 0; o >>= 1) {
        if (t < o) { r1[t] += r1[t + o]; r2[t] += r2[t + o]; }
        __syncthreads();
    }
    float m = r1[0] * (1.f / 64.f), var = r2[0] * (1.f / 64.f) - m * m;
    out[(size_t)row * 64 + t] = (_Float16)((x - m) * rsqrtf(var + LN_EPSF) * g[t] + bb[t]);
}

// ============================ slot attention ===============================
__global__ void slot_q_kernel(const float* __restrict__ slots,
                              const float* __restrict__ g, const float* __restrict__ bb,
                              const float* __restrict__ wq, const float* __restrict__ bq,
                              float* __restrict__ Q)
{
    int row = blockIdx.x, t = threadIdx.x;   // 28 rows x 64 threads
    __shared__ float s[64], r1[64], r2[64];
    float x = slots[row * 64 + t];
    r1[t] = x; r2[t] = x * x; __syncthreads();
    for (int o = 32; o > 0; o >>= 1) {
        if (t < o) { r1[t] += r1[t + o]; r2[t] += r2[t + o]; }
        __syncthreads();
    }
    float m = r1[0] * (1.f / 64.f), var = r2[0] * (1.f / 64.f) - m * m;
    s[t] = (x - m) * rsqrtf(var + LN_EPSF) * g[t] + bb[t];
    __syncthreads();
    float acc = bq[t];
    for (int d = 0; d < 64; ++d) acc += s[d] * wq[t * 64 + d];
    Q[row * 64 + t] = acc;
}

__global__ void attn_kernel(const float* __restrict__ Q,
                            const _Float16* __restrict__ K, const _Float16* __restrict__ V,
                            float* __restrict__ UPD, float* __restrict__ SUMA)
{
    const int b = blockIdx.y, tid = threadIdx.x;
    const int j = blockIdx.x * 256 + tid;    // 49*256 == 12544 exactly
    __shared__ float qs[448], pu[448], ps[7];
    if (tid < 448) { qs[tid] = Q[b * 448 + tid]; pu[tid] = 0.f; }
    if (tid < 7) ps[tid] = 0.f;
    __syncthreads();
    const _Float16* krow = K + ((size_t)b * 12544 + j) * 64;
    float dots[7]; for (int i = 0; i < 7; ++i) dots[i] = 0.f;
    for (int d = 0; d < 64; ++d) {
        float kv = (float)krow[d];
        for (int i = 0; i < 7; ++i) dots[i] += qs[i * 64 + d] * kv;
    }
    float mx = -1e30f;
    for (int i = 0; i < 7; ++i) { dots[i] *= 0.125f; mx = fmaxf(mx, dots[i]); }
    float se = 0.f, p[7];
    for (int i = 0; i < 7; ++i) { p[i] = __expf(dots[i] - mx); se += p[i]; }
    float inv = 1.f / se;
    for (int i = 0; i < 7; ++i) { p[i] = p[i] * inv + 1e-8f; atomicAdd(&ps[i], p[i]); }
    const _Float16* vrow = V + ((size_t)b * 12544 + j) * 64;
    for (int d = 0; d < 64; ++d) {
        float vv = (float)vrow[d];
        for (int i = 0; i < 7; ++i) atomicAdd(&pu[i * 64 + d], p[i] * vv);
    }
    __syncthreads();
    if (tid < 448) atomicAdd(&UPD[b * 448 + tid], pu[tid]);
    if (tid < 7)   atomicAdd(&SUMA[b * 7 + tid], ps[tid]);
}

__global__ void slot_update_kernel(float* __restrict__ slots,
                                   const float* __restrict__ UPD, const float* __restrict__ SUMA,
                                   const float* __restrict__ w_ih, const float* __restrict__ w_hh,
                                   const float* __restrict__ b_ih, const float* __restrict__ b_hh,
                                   const float* __restrict__ lng, const float* __restrict__ lnb,
                                   const float* __restrict__ w1, const float* __restrict__ b1,
                                   const float* __restrict__ w2, const float* __restrict__ b2)
{
    int row = blockIdx.x, t = threadIdx.x;   // 28 rows x 64 threads
    __shared__ float u[64], h[64], s[64], r1[64], r2[64], hd[64], pre[64];
    u[t] = UPD[row * 64 + t] / SUMA[row];
    h[t] = slots[row * 64 + t];
    __syncthreads();
    float gi[3], gh[3];
    for (int k2 = 0; k2 < 3; ++k2) {
        float a = b_ih[k2 * 64 + t], c = b_hh[k2 * 64 + t];
        const float* wi = w_ih + (size_t)(k2 * 64 + t) * 64;
        const float* wh = w_hh + (size_t)(k2 * 64 + t) * 64;
        for (int d = 0; d < 64; ++d) { a += u[d] * wi[d]; c += h[d] * wh[d]; }
        gi[k2] = a; gh[k2] = c;
    }
    float r = 1.f / (1.f + __expf(-(gi[0] + gh[0])));
    float z = 1.f / (1.f + __expf(-(gi[1] + gh[1])));
    float n = tanhf(gi[2] + r * gh[2]);
    float nv = (1.f - z) * n + z * h[t];
    s[t] = nv; r1[t] = nv; r2[t] = nv * nv;
    __syncthreads();
    for (int o = 32; o > 0; o >>= 1) {
        if (t < o) { r1[t] += r1[t + o]; r2[t] += r2[t + o]; }
        __syncthreads();
    }
    float m = r1[0] * (1.f / 64.f), var = r2[0] * (1.f / 64.f) - m * m;
    pre[t] = (nv - m) * rsqrtf(var + LN_EPSF) * lng[t] + lnb[t];
    __syncthreads();
    float a = b1[t];
    for (int d = 0; d < 64; ++d) a += pre[d] * w1[t * 64 + d];
    hd[t] = fmaxf(a, 0.f);
    __syncthreads();
    float o2 = b2[t];
    for (int d = 0; d < 64; ++d) o2 += hd[d] * w2[t * 64 + d];
    slots[row * 64 + t] = nv + o2;
}

// =============================== decoder ===================================
__global__ void dec_init_kernel(const float* __restrict__ slots,
                                const float* __restrict__ w, const float* __restrict__ bb,
                                _Float16* __restrict__ out)
{
    const int total = 28 * 8 * 8 * 64;
    int idx = blockIdx.x * blockDim.x + threadIdx.x;
    if (idx >= total) return;
    int d = idx & 63;
    int p = idx >> 6;              // img*64 + y*8 + x
    int img = p >> 6, yx = p & 63;
    int y = yx >> 3, x = yx & 7;
    float gy = y * (1.f / 7.f), gx = x * (1.f / 7.f);
    float pos = w[d * 4] * gy + w[d * 4 + 1] * gx +
                w[d * 4 + 2] * (1.f - gy) + w[d * 4 + 3] * (1.f - gx) + bb[d];
    out[idx] = (_Float16)(slots[img * 64 + d] + pos);
}

// ================================ utils ====================================
__global__ void zero_f32_kernel(float* p, int n)
{
    for (int i = blockIdx.x * blockDim.x + threadIdx.x; i < n; i += gridDim.x * blockDim.x)
        p[i] = 0.f;
}
__global__ void copy_f32_kernel(float* d, const float* s, int n)
{
    for (int i = blockIdx.x * blockDim.x + threadIdx.x; i < n; i += gridDim.x * blockDim.x)
        d[i] = s[i];
}

// ================================ host =====================================
struct ParamsMap {
    const float *image, *init_slots;
    const float *enc_w[4], *enc_b[4], *encpos_w, *encpos_b;
    const float *fc1_w, *fc1_b, *fc2_w, *fc2_b;
    const float *toq_w, *toq_b, *tok_w, *tok_b, *tov_w, *tov_b;
    const float *gru_wih, *gru_whh, *gru_bih, *gru_bhh;
    const float *fc1s_w, *fc1s_b, *fc2s_w, *fc2s_b;
    const float *lnin_g, *lnin_b, *lns_g, *lns_b, *lnp_g, *lnp_b;
    const float *decpos_w, *decpos_b;
    const float *dec_w[6], *dec_b[6];
};

extern "C" void kernel_launch(void* const* d_in, const int* in_sizes, int n_in,
                              void* d_out, int out_size, void* d_ws, size_t ws_size,
                              hipStream_t stream)
{
    (void)out_size; (void)ws_size; (void)n_in;
    ParamsMap P;
    #define F(i) ((const float*)d_in[i])
    P.image = F(0); P.init_slots = F(1);
    if (in_sizes[2] == 64) {
        // JAX tree_leaves order: dicts flattened with sorted keys
        P.dec_b[0]=F(2);  P.dec_w[0]=F(3);  P.dec_b[1]=F(4);  P.dec_w[1]=F(5);
        P.dec_b[2]=F(6);  P.dec_w[2]=F(7);  P.dec_b[3]=F(8);  P.dec_w[3]=F(9);
        P.dec_b[4]=F(10); P.dec_w[4]=F(11); P.dec_b[5]=F(12); P.dec_w[5]=F(13);
        P.decpos_b=F(14); P.decpos_w=F(15);
        P.enc_b[0]=F(16); P.enc_w[0]=F(17); P.enc_b[1]=F(18); P.enc_w[1]=F(19);
        P.enc_b[2]=F(20); P.enc_w[2]=F(21); P.enc_b[3]=F(22); P.enc_w[3]=F(23);
        P.encpos_b=F(24); P.encpos_w=F(25);
        P.fc1_b=F(26);  P.fc1_w=F(27);  P.fc1s_b=F(28); P.fc1s_w=F(29);
        P.fc2_b=F(30);  P.fc2_w=F(31);  P.fc2s_b=F(32); P.fc2s_w=F(33);
        P.gru_bhh=F(34); P.gru_bih=F(35); P.gru_whh=F(36); P.gru_wih=F(37);
        P.lnin_b=F(38); P.lnin_g=F(39); P.lnp_b=F(40); P.lnp_g=F(41);
        P.lns_b=F(42);  P.lns_g=F(43);
        P.tok_b=F(44); P.tok_w=F(45); P.toq_b=F(46); P.toq_w=F(47);
        P.tov_b=F(48); P.tov_w=F(49);
    } else {
        // dict insertion order, w before b as constructed
        P.enc_w[0]=F(2);  P.enc_b[0]=F(3);  P.enc_w[1]=F(4);  P.enc_b[1]=F(5);
        P.enc_w[2]=F(6);  P.enc_b[2]=F(7);  P.enc_w[3]=F(8);  P.enc_b[3]=F(9);
        P.encpos_w=F(10); P.encpos_b=F(11);
        P.fc1_w=F(12); P.fc1_b=F(13); P.fc2_w=F(14); P.fc2_b=F(15);
        P.toq_w=F(16); P.toq_b=F(17); P.tok_w=F(18); P.tok_b=F(19);
        P.tov_w=F(20); P.tov_b=F(21);
        P.gru_wih=F(22); P.gru_whh=F(23); P.gru_bih=F(24); P.gru_bhh=F(25);
        P.fc1s_w=F(26); P.fc1s_b=F(27); P.fc2s_w=F(28); P.fc2s_b=F(29);
        P.lnin_g=F(30); P.lnin_b=F(31); P.lns_g=F(32); P.lns_b=F(33);
        P.lnp_g=F(34);  P.lnp_b=F(35);
        P.decpos_w=F(36); P.decpos_b=F(37);
        P.dec_w[0]=F(38); P.dec_b[0]=F(39); P.dec_w[1]=F(40); P.dec_b[1]=F(41);
        P.dec_w[2]=F(42); P.dec_b[2]=F(43); P.dec_w[3]=F(44); P.dec_b[3]=F(45);
        P.dec_w[4]=F(46); P.dec_b[4]=F(47); P.dec_w[5]=F(48); P.dec_b[5]=F(49);
    }
    #undef F

    char* ws = (char*)d_ws;
    _Float16* WP   = (_Float16*)ws;                       // packed f16 weights
    _Float16* ACT0 = (_Float16*)(ws + 1894400ull);        // 58.7MB activation ping
    _Float16* ACT1 = (_Float16*)(ws + 60614656ull);       // 58.7MB activation pong
    _Float16* KB   = (_Float16*)(ws + 119334912ull);      // k  (4,12544,64) f16
    _Float16* VB   = (_Float16*)(ws + 125757440ull);      // v
    float* Q     = (float*)(ws + 132179968ull);
    float* SLOTS = (float*)(ws + 132187136ull);
    float* UPD   = (float*)(ws + 132194304ull);           // followed by SUMA
    float* SUMA  = (float*)(ws + 132201472ull);
    float* STATS = (float*)(ws + 132201600ull);
    float* out   = (float*)d_out;

    // packed-weight half offsets
    const size_t wENC[4] = {0, 102400, 204800, 307200};
    const size_t wDEC[6] = {409600, 512000, 614400, 716800, 819200, 921600};
    const size_t wFC1 = 930816, wFC2 = 934912, wTOK = 939008, wTOV = 943104;

    auto packw = [&](const float* src, size_t wof, int O, int I, int K, int nt) {
        int total = K * K * 2 * nt * 512;
        pack_w_kernel<<<(total + 255) / 256, 256, 0, stream>>>(src, WP + wof, O, I, K, nt);
    };
    auto conv = [&](const _Float16* in, _Float16* o16, float* f32o, size_t wof,
                    const float* bias, int Ni, int iH, int iW, int oH, int oW,
                    int K, int S, int pad, int Cout, int nt, int relu) {
        int tiles = (Ni * oH * oW + 15) / 16;
        int blocks = (tiles + 3) / 4;
        conv_wmma_kernel<<<blocks, 128, 0, stream>>>(in, WP + wof, bias, o16, f32o,
                                                     Ni, iH, iW, oH, oW, K, S, pad, Cout, nt, relu);
    };

    // ---- pack all GEMM weights to WMMA B-fragment layout (f16) ----
    packw(P.enc_w[0], wENC[0], 64, 3, 5, 4);
    packw(P.enc_w[1], wENC[1], 64, 64, 5, 4);
    packw(P.enc_w[2], wENC[2], 64, 64, 5, 4);
    packw(P.enc_w[3], wENC[3], 64, 64, 5, 4);
    for (int i = 0; i < 5; ++i) packw(P.dec_w[i], wDEC[i], 64, 64, 5, 4);
    packw(P.dec_w[5], wDEC[5], 4, 64, 3, 1);
    packw(P.fc1_w, wFC1, 64, 64, 1, 4);
    packw(P.fc2_w, wFC2, 64, 64, 1, 4);
    packw(P.tok_w, wTOK, 64, 64, 1, 4);
    packw(P.tov_w, wTOV, 64, 64, 1, 4);

    // ---- encoder ----
    pack_image_kernel<<<(3211264 + 255) / 256, 256, 0, stream>>>(P.image, ACT0);
    conv(ACT0, ACT1, nullptr, wENC[0], P.enc_b[0], 4, 112, 112, 112, 112, 5, 1, 2, 64, 4, 1);
    conv(ACT1, ACT0, nullptr, wENC[1], P.enc_b[1], 4, 112, 112, 112, 112, 5, 1, 2, 64, 4, 1);
    conv(ACT0, ACT1, nullptr, wENC[2], P.enc_b[2], 4, 112, 112, 112, 112, 5, 1, 2, 64, 4, 1);
    conv(ACT1, ACT0, nullptr, wENC[3], P.enc_b[3], 4, 112, 112, 112, 112, 5, 1, 2, 64, 4, 1);
    add_pos_kernel<<<(3211264 + 255) / 256, 256, 0, stream>>>(ACT0, P.encpos_w, P.encpos_b);
    zero_f32_kernel<<<1, 256, 0, stream>>>(STATS, 8);
    gstats_kernel<<<256, 256, 0, stream>>>(ACT0, STATS);
    gnorm_kernel<<<(3211264 + 255) / 256, 256, 0, stream>>>(ACT0, STATS);
    conv(ACT0, ACT1, nullptr, wFC1, P.fc1_b, 4, 112, 112, 112, 112, 1, 1, 0, 64, 4, 1);
    conv(ACT1, ACT0, nullptr, wFC2, P.fc2_b, 4, 112, 112, 112, 112, 1, 1, 0, 64, 4, 0);
    ln_rows_kernel<<<50176, 64, 0, stream>>>(ACT0, ACT1, P.lnin_g, P.lnin_b);
    conv(ACT1, KB, nullptr, wTOK, P.tok_b, 4, 112, 112, 112, 112, 1, 1, 0, 64, 4, 0);
    conv(ACT1, VB, nullptr, wTOV, P.tov_b, 4, 112, 112, 112, 112, 1, 1, 0, 64, 4, 0);

    // ---- slot attention (3 iterations) ----
    copy_f32_kernel<<<7, 256, 0, stream>>>(SLOTS, P.init_slots, 1792);
    for (int it = 0; it < 3; ++it) {
        slot_q_kernel<<<28, 64, 0, stream>>>(SLOTS, P.lns_g, P.lns_b, P.toq_w, P.toq_b, Q);
        zero_f32_kernel<<<8, 256, 0, stream>>>(UPD, 1820);   // UPD + SUMA contiguous
        attn_kernel<<<dim3(49, 4), 256, 0, stream>>>(Q, KB, VB, UPD, SUMA);
        slot_update_kernel<<<28, 64, 0, stream>>>(SLOTS, UPD, SUMA,
            P.gru_wih, P.gru_whh, P.gru_bih, P.gru_bhh,
            P.lnp_g, P.lnp_b, P.fc1s_w, P.fc1s_b, P.fc2s_w, P.fc2s_b);
    }
    copy_f32_kernel<<<7, 256, 0, stream>>>(out + 1404928, SLOTS, 1792);  // slots output

    // ---- decoder ----
    dec_init_kernel<<<(114688 + 255) / 256, 256, 0, stream>>>(SLOTS, P.decpos_w, P.decpos_b, ACT0);
    conv(ACT0, ACT1, nullptr, wDEC[0], P.dec_b[0], 28, 8, 8, 16, 16, 5, 2, 2, 64, 4, 1);
    conv(ACT1, ACT0, nullptr, wDEC[1], P.dec_b[1], 28, 16, 16, 32, 32, 5, 2, 2, 64, 4, 1);
    conv(ACT0, ACT1, nullptr, wDEC[2], P.dec_b[2], 28, 32, 32, 64, 64, 5, 2, 2, 64, 4, 1);
    conv(ACT1, ACT0, nullptr, wDEC[3], P.dec_b[3], 28, 64, 64, 128, 128, 5, 2, 2, 64, 4, 1);
    conv(ACT0, ACT1, nullptr, wDEC[4], P.dec_b[4], 28, 128, 128, 128, 128, 5, 1, 2, 64, 4, 1);
    // final 3x3 conv: Cout=4, crop to 112x112, writes recons+masks into d_out
    conv(ACT1, nullptr, out, wDEC[5], P.dec_b[5], 28, 128, 128, 128, 128, 3, 1, 1, 4, 1, 0);
}
